// MyBertSelfAttention_19533511262281
// MI455X (gfx1250) — compile-verified, matrix operand order checked by reading, not run on hardware
//
#include <hip/hip_runtime.h>

typedef __attribute__((ext_vector_type(16))) _Float16 v16h;
typedef __attribute__((ext_vector_type(8)))  _Float16 v8h;
typedef __attribute__((ext_vector_type(8)))  float    v8f;
typedef __attribute__((ext_vector_type(4)))  float    v4f;

#define HIDDEN 768
#define MROWS  (8 * 4096)          // 32768 rows (B*S)
#define NOUT   (3 * HIDDEN)        // 2304 output cols (q|k|v)
#define KC     32                  // K chunk = one WMMA K
#define KITER  (HIDDEN / KC)       // 24
#define MT     128                 // block M tile
#define NT     64                  // block N tile
#define LDSA   40                  // padded LDS stride (halves) -> conflict-free frag reads
#define LDSB   40

__global__ __launch_bounds__(256)
void qkv_wmma_kernel(const float* __restrict__ hs,
                     const float* __restrict__ wq, const float* __restrict__ qb,
                     const float* __restrict__ wk, const float* __restrict__ kb,
                     const float* __restrict__ wv, const float* __restrict__ vb,
                     float* __restrict__ out)
{
    __shared__ _Float16 As[2][MT][LDSA];
    __shared__ _Float16 Bs[2][NT][LDSB];

    const int t    = threadIdx.x;
    const int lane = t & 31;
    const int wave = t >> 5;
    const int wm   = wave >> 1;           // 0..3 : wave M position
    const int wn   = wave & 1;            // 0..1 : wave N position

    const int mBase = blockIdx.y * MT;
    const int nBase = blockIdx.x * NT;
    const int seg   = nBase / HIDDEN;     // whole tile lives in one of q/k/v
    const int nSeg  = nBase % HIDDEN;

    const float* __restrict__ W  = (seg == 0) ? wq : (seg == 1) ? wk : wv;
    const float* __restrict__ Bi = (seg == 0) ? qb : (seg == 1) ? kb : vb;
    float* __restrict__ O = out + (size_t)seg * (size_t)MROWS * HIDDEN;

    // ---- global->LDS staging assignments (f32 load, f16 convert) ----
    // A: 128 rows x 32 k; 2 threads per row, 16 floats each
    const int aRow  = t >> 1;
    const int aHalf = t & 1;
    const float* gA = hs + (size_t)(mBase + aRow) * HIDDEN + aHalf * 16;
    // B: 64 rows (n) x 32 k; 4 threads per row, 8 floats each
    const int wRow  = t >> 2;
    const int wQuad = t & 3;
    const float* gB = W + (size_t)(nSeg + wRow) * HIDDEN + wQuad * 8;

    auto stage = [&](int buf, int k0) {
        v4f a0 = *(const v4f*)(gA + k0 + 0);
        v4f a1 = *(const v4f*)(gA + k0 + 4);
        v4f a2 = *(const v4f*)(gA + k0 + 8);
        v4f a3 = *(const v4f*)(gA + k0 + 12);
        v16h av;
#pragma unroll
        for (int i = 0; i < 4; ++i) {
            av[i]      = (_Float16)a0[i];
            av[4 + i]  = (_Float16)a1[i];
            av[8 + i]  = (_Float16)a2[i];
            av[12 + i] = (_Float16)a3[i];
        }
        *(v16h*)&As[buf][aRow][aHalf * 16] = av;

        v4f b0 = *(const v4f*)(gB + k0 + 0);
        v4f b1 = *(const v4f*)(gB + k0 + 4);
        v8h bvv;
#pragma unroll
        for (int i = 0; i < 4; ++i) {
            bvv[i]     = (_Float16)b0[i];
            bvv[4 + i] = (_Float16)b1[i];
        }
        *(v8h*)&Bs[buf][wRow][wQuad * 8] = bvv;
    };

    v8f acc[2][2] = {};

    stage(0, 0);
    __syncthreads();

    const int ko   = (lane >> 4) * 16;   // per-lane K offset inside chunk
    const int lrow = lane & 15;          // per-lane M (A) / N (B) index

    int buf = 0;
#pragma unroll 1
    for (int kc = 0; kc < KITER; ++kc) {
        const int k0 = kc * KC;
        if (kc + 2 < KITER) {
            __builtin_prefetch((const void*)(gA + k0 + 2 * KC), 0, 1);
            __builtin_prefetch((const void*)(gB + k0 + 2 * KC), 0, 1);
        }
        if (kc + 1 < KITER) stage(buf ^ 1, k0 + KC);

        v16h af0 = *(const v16h*)&As[buf][wm * 32 + lrow][ko];
        v16h af1 = *(const v16h*)&As[buf][wm * 32 + 16 + lrow][ko];
        v16h bf0 = *(const v16h*)&Bs[buf][wn * 32 + lrow][ko];
        v16h bf1 = *(const v16h*)&Bs[buf][wn * 32 + 16 + lrow][ko];

        acc[0][0] = __builtin_amdgcn_wmma_f32_16x16x32_f16(false, af0, false, bf0,
                                                           (short)0, acc[0][0], false, false);
        acc[0][1] = __builtin_amdgcn_wmma_f32_16x16x32_f16(false, af0, false, bf1,
                                                           (short)0, acc[0][1], false, false);
        acc[1][0] = __builtin_amdgcn_wmma_f32_16x16x32_f16(false, af1, false, bf0,
                                                           (short)0, acc[1][0], false, false);
        acc[1][1] = __builtin_amdgcn_wmma_f32_16x16x32_f16(false, af1, false, bf1,
                                                           (short)0, acc[1][1], false, false);

        __syncthreads();
        buf ^= 1;
    }

    // ---- epilogue: bias add + store (C/D layout: lane L, vgpr r -> m = r + 8*(L/16), n = L%16) ----
#pragma unroll
    for (int mi = 0; mi < 2; ++mi) {
#pragma unroll
        for (int ni = 0; ni < 2; ++ni) {
            const int col   = nSeg + wn * 32 + ni * 16 + (lane & 15);
            const float bias = Bi[col];
            const int mrow0 = mBase + wm * 32 + mi * 16 + 8 * (lane >> 4);
            float* op = O + (size_t)mrow0 * HIDDEN + col;
#pragma unroll
            for (int r = 0; r < 8; ++r)
                op[(size_t)r * HIDDEN] = acc[mi][ni][r] + bias;
        }
    }
}

extern "C" void kernel_launch(void* const* d_in, const int* in_sizes, int n_in,
                              void* d_out, int out_size, void* d_ws, size_t ws_size,
                              hipStream_t stream) {
    (void)in_sizes; (void)n_in; (void)out_size; (void)d_ws; (void)ws_size;
    const float* hs = (const float*)d_in[0];
    const float* wq = (const float*)d_in[1];
    const float* qb = (const float*)d_in[2];
    const float* wk = (const float*)d_in[3];
    const float* kb = (const float*)d_in[4];
    const float* wv = (const float*)d_in[5];
    const float* vb = (const float*)d_in[6];
    float* out = (float*)d_out;

    dim3 grid(NOUT / NT, MROWS / MT);   // 36 x 256 blocks
    dim3 block(256);
    qkv_wmma_kernel<<<grid, block, 0, stream>>>(hs, wq, qb, wk, kb, wv, vb, out);
}
